// MozafariMNIST2018_36026185679107
// MI455X (gfx1250) — matrix operand order repaired and checked by reading, not compiled, not run on hardware
//
#include <hip/hip_runtime.h>

typedef __attribute__((ext_vector_type(16))) _Float16 v16h;
typedef __attribute__((ext_vector_type(8)))  _Float16 v8h;
typedef __attribute__((ext_vector_type(8)))  float    v8f;

#define T_STEPS 15
#define C1      6
#define HIN     160
#define F1      30
#define HP      80            // pooled spatial
#define F2      250
#define KDIM    270           // 30*3*3
#define KPAD    288           // 9 k-steps of 32
#define MPAD    256           // 250 -> 16 tiles of 16
#define NTILE   32            // output pixels per block (2 n-tiles)
#define THR1    15.0f
#define THR2    10.0f
#define NSPAT   (HP*HP)                 // 6400
#define NTOT    (T_STEPS*NSPAT)         // 96000
#define ELEMS   (T_STEPS*F2*NSPAT)      // 24,000,000
#define FHW     (F2*NSPAT)              // 1,600,000

// ---------------------------------------------------------------------------
// k0: pack w2 (250,30,3,3) f32 -> f16 [256][288], zero-pad; init atomic max.
// ---------------------------------------------------------------------------
__global__ void k0_prep(const float* __restrict__ w2, _Float16* __restrict__ w2h,
                        unsigned* __restrict__ vmax) {
  int idx = blockIdx.x * 256 + threadIdx.x;
  if (idx == 0) *vmax = 0u;
  if (idx >= MPAD * KPAD) return;
  int f = idx / KPAD, k = idx % KPAD;
  float v = (f < F2 && k < KDIM) ? w2[f * KDIM + k] : 0.0f;
  w2h[idx] = (_Float16)v;
}

// ---------------------------------------------------------------------------
// k1: conv1 (pad 2, 5x5) + fire(15) + maxpool2x2 fused.
// Pool of binary spikes == OR of (pot >= 15) over the 2x2 raw window.
// One block = 16x16 pooled tile for one t; loops over 30 features.
// ---------------------------------------------------------------------------
__global__ void k1_conv1_fire_pool(const int* __restrict__ in,
                                   const float* __restrict__ w1,
                                   _Float16* __restrict__ spk) {
  __shared__ float sIn[C1 * 36 * 36];   // 36x36 raw region (32 raw + 4 halo)
  __shared__ float sW[F1 * 150];
  int bid = blockIdx.x;
  int t = bid / 25, tile = bid % 25;
  int py0 = (tile / 5) * 16, px0 = (tile % 5) * 16;
  int tid = threadIdx.x;

  for (int idx = tid; idx < C1 * 36 * 36; idx += 256) {
    int c = idx / 1296, rr = (idx / 36) % 36, cc = idx % 36;
    int gy = 2 * py0 - 2 + rr, gx = 2 * px0 - 2 + cc;
    float v = 0.0f;
    if (gy >= 0 && gy < HIN && gx >= 0 && gx < HIN)
      v = (float)in[((t * C1 + c) * HIN + gy) * HIN + gx];
    sIn[idx] = v;
  }
  for (int idx = tid; idx < F1 * 150; idx += 256) sW[idx] = w1[idx];
  __syncthreads();

  int ly = tid >> 4, lx = tid & 15;
  for (int f = 0; f < F1; ++f) {
    float p00 = 0, p01 = 0, p10 = 0, p11 = 0;
    const float* wf = &sW[f * 150];
    for (int c = 0; c < C1; ++c)
      for (int r = 0; r < 5; ++r) {
        int rb = c * 1296 + (2 * ly + r) * 36 + 2 * lx;
        const float* wr = &wf[c * 25 + r * 5];
#pragma unroll
        for (int s = 0; s < 5; ++s) {
          float w = wr[s];
          p00 += w * sIn[rb + s];
          p01 += w * sIn[rb + s + 1];
          p10 += w * sIn[rb + s + 36];
          p11 += w * sIn[rb + s + 37];
        }
      }
    bool sp = (p00 >= THR1) | (p01 >= THR1) | (p10 >= THR1) | (p11 >= THR1);
    spk[((t * F1 + f) * HP + py0 + ly) * HP + px0 + lx] =
        sp ? (_Float16)1.0f : (_Float16)0.0f;
  }
}

// ---------------------------------------------------------------------------
// k2: conv2 as implicit GEMM with V_WMMA_F32_16X16X32_F16.
// GEMM: D[256pad x N] = W[256x288] * im2col[288 x N], N = 96000.
// Block: 32 output pixels (2 n-tiles), im2col 32x288 f16 tile in LDS (18 KB).
// 8 waves x (2 f-tiles x 2 n-tiles) x 9 k-steps = 288 WMMAs/block; each A
// and B fragment feeds two WMMAs. Fused fire(10) store.
// A fragment (16x32): lane<16 rows M, K 0-15;
// lane>=16 same rows, K 16-31. B fragment: lane -> col (lane&15),
// 16 consecutive K starting at (lane>>4)*16.
// ---------------------------------------------------------------------------
__global__ void k2_conv2_wmma(const _Float16* __restrict__ spk,
                              const _Float16* __restrict__ w2h,
                              float* __restrict__ pot) {
  __shared__ __align__(32) _Float16 sB[NTILE * KPAD];   // im2col [n][k]
  int n0 = blockIdx.x * NTILE;
  int tid = threadIdx.x;

  // Build im2col tile; each column decodes its own (t,y,x), pad-1 by bounds.
  for (int idx = tid; idx < NTILE * KPAD; idx += 256) {
    int n = idx / KPAD, k = idx % KPAD;
    _Float16 v = (_Float16)0.0f;
    if (k < KDIM) {
      int g = n0 + n;
      int t = g / NSPAT;
      int rem = g % NSPAT;
      int y = rem / HP, x = rem % HP;
      int c = k / 9, r = (k % 9) / 3, s = k % 3;
      int yy = y - 1 + r, xx = x - 1 + s;
      if (yy >= 0 && yy < HP && xx >= 0 && xx < HP)
        v = spk[((t * F1 + c) * HP + yy) * HP + xx];
    }
    sB[idx] = v;
  }
  __syncthreads();

  int wv = tid >> 5, lane = tid & 31;
  int MB0 = wv * 32, MB1 = MB0 + 16;        // two f-tiles per wave
  int row = lane & 15;
  int hi = lane >> 4;
  int ncol = lane & 15;
  v8f acc00 = {}, acc01 = {}, acc10 = {}, acc11 = {};

  // Warm the L2/WGP$ path for this wave's weight rows (global_prefetch_b8).
  __builtin_prefetch(&w2h[(MB0 + row) * KPAD], 0, 3);
  __builtin_prefetch(&w2h[(MB1 + row) * KPAD], 0, 3);

#pragma unroll
  for (int kb = 0; kb < KPAD / 32; ++kb) {
    int KB = kb * 32;
    // B fragments for the two n-tiles: 16 consecutive halves, 32B aligned.
    v16h b0 = *(const v16h*)&sB[ncol * KPAD + KB + hi * 16];
    v16h b1 = *(const v16h*)&sB[(16 + ncol) * KPAD + KB + hi * 16];
    // A fragments for the two f-tiles: two 16B global loads each.
    int ka = KB + hi * 16;
    const v8h* a0p = (const v8h*)&w2h[(MB0 + row) * KPAD + ka];
    const v8h* a1p = (const v8h*)&w2h[(MB1 + row) * KPAD + ka];
    v8h a0lo = a0p[0], a0hi = a0p[1];   // +1 v8h == +8 halves
    v8h a1lo = a1p[0], a1hi = a1p[1];
    v16h a0, a1;
#pragma unroll
    for (int i = 0; i < 8; ++i) {
      a0[i] = a0lo[i]; a0[i + 8] = a0hi[i];
      a1[i] = a1lo[i]; a1[i + 8] = a1hi[i];
    }
    acc00 = __builtin_amdgcn_wmma_f32_16x16x32_f16(false, a0, false, b0,
                                                   (short)0, acc00, false, false);
    acc01 = __builtin_amdgcn_wmma_f32_16x16x32_f16(false, a0, false, b1,
                                                   (short)0, acc01, false, false);
    acc10 = __builtin_amdgcn_wmma_f32_16x16x32_f16(false, a1, false, b0,
                                                   (short)0, acc10, false, false);
    acc11 = __builtin_amdgcn_wmma_f32_16x16x32_f16(false, a1, false, b1,
                                                   (short)0, acc11, false, false);
  }

  // D layout: VGPR r -> (M = r + 8*hi, N = lane&15). Fused fire(THR2).
  int g0 = n0 + ncol, g1 = g0 + 16;
  int t0 = g0 / NSPAT, hw0 = g0 % NSPAT;
  int t1 = g1 / NSPAT, hw1 = g1 % NSPAT;
#pragma unroll
  for (int rr = 0; rr < 8; ++rr) {
    int m = rr + hi * 8;
    int f0 = MB0 + m, f1 = MB1 + m;
    float v;
    v = acc00[rr]; v = (v < THR2) ? 0.0f : v;
    if (f0 < F2) pot[((size_t)t0 * F2 + f0) * NSPAT + hw0] = v;
    v = acc01[rr]; v = (v < THR2) ? 0.0f : v;
    if (f0 < F2) pot[((size_t)t1 * F2 + f0) * NSPAT + hw1] = v;
    v = acc10[rr]; v = (v < THR2) ? 0.0f : v;
    if (f1 < F2) pot[((size_t)t0 * F2 + f1) * NSPAT + hw0] = v;
    v = acc11[rr]; v = (v < THR2) ? 0.0f : v;
    if (f1 < F2) pot[((size_t)t1 * F2 + f1) * NSPAT + hw1] = v;
  }
}

// ---------------------------------------------------------------------------
// k3: per (t,h,w) max + argmax over 250 features (wave32 shuffle reduction,
// first-max tie rule matching jnp.argmax).
// ---------------------------------------------------------------------------
__global__ void k3_maxarg(const float* __restrict__ pot,
                          int* __restrict__ maxidx, int* __restrict__ clampf) {
  int gid = blockIdx.x * 8 + (threadIdx.x >> 5);
  if (gid >= NTOT) return;
  int lane = threadIdx.x & 31;
  int t = gid / NSPAT, hw = gid % NSPAT;
  const float* base = pot + (size_t)t * F2 * NSPAT + hw;
  float bv = -1.0f; int bi = 0;
#pragma unroll
  for (int j = 0; j < 8; ++j) {
    int f = lane + j * 32;
    if (f < F2) {
      float v = base[(size_t)f * NSPAT];
      if (v > bv) { bv = v; bi = f; }
    }
  }
#pragma unroll
  for (int off = 16; off >= 1; off >>= 1) {
    float ov = __shfl_xor(bv, off, 32);
    int   oi = __shfl_xor(bi, off, 32);
    if (ov > bv || (ov == bv && oi < bi)) { bv = ov; bi = oi; }
  }
  if (lane == 0) { maxidx[gid] = bi; clampf[gid] = (bv > 0.0f) ? 1 : 0; }
}

// k3b: earliest spike time per (h,w) -> winner channel, gated by last step.
__global__ void k3b_winch(const int* __restrict__ maxidx,
                          const int* __restrict__ clampf,
                          int* __restrict__ winch) {
  int hw = blockIdx.x * 256 + threadIdx.x;
  if (hw >= NSPAT) return;
  int s = 0;
  for (int t = 0; t < T_STEPS; ++t) s += clampf[t * NSPAT + hw];
  int e = T_STEPS - s;
  if (e < 0) e = 0;
  if (e > T_STEPS - 1) e = T_STEPS - 1;
  int win  = maxidx[e * NSPAT + hw];
  int gate = clampf[(T_STEPS - 1) * NSPAT + hw];
  winch[hw] = gate ? win : -1;
}

// k4: apply pointwise inhibition in place; spk = sign(pot).
__global__ void k4_apply(float* __restrict__ pot, float* __restrict__ spk,
                         const int* __restrict__ winch) {
  size_t idx = (size_t)blockIdx.x * 256 + threadIdx.x;
  if (idx >= (size_t)ELEMS) return;
  int hw = (int)(idx % NSPAT);
  int f  = (int)((idx / NSPAT) % F2);
  float p = pot[idx];
  float o = (winch[hw] == f) ? p : 0.0f;
  pot[idx] = o;
  spk[idx] = (o > 0.0f) ? 1.0f : 0.0f;
}

// k5: per (f,h,w): spike count, value at first spike time, global max(trunc).
__global__ void k5_stats(const float* __restrict__ pot,
                         const float* __restrict__ spk,
                         float* __restrict__ valA, int* __restrict__ nspkA,
                         unsigned* __restrict__ vmax) {
  int i = blockIdx.x * 256 + threadIdx.x;
  if (i >= FHW) return;
  int f = i / NSPAT, hw = i % NSPAT;
  int ns = 0;
  for (int t = 0; t < T_STEPS; ++t)
    ns += (spk[((size_t)t * F2 + f) * NSPAT + hw] > 0.0f) ? 1 : 0;
  int first = T_STEPS - ns;
  if (first < 0) first = 0;
  if (first > T_STEPS - 1) first = T_STEPS - 1;
  float val = pot[((size_t)first * F2 + f) * NSPAT + hw];
  valA[i] = val;
  nspkA[i] = ns;
  float masked = (ns > 0) ? val : 0.0f;  // nonnegative -> uint order == float order
  atomicMax(vmax, __float_as_uint(masked));
}

// k6: total = nspk * (value + v), v = max(trunc) * T.
__global__ void k6_total(const float* __restrict__ valA,
                         const int* __restrict__ nspkA,
                         const unsigned* __restrict__ vmax,
                         float* __restrict__ total) {
  int i = blockIdx.x * 256 + threadIdx.x;
  if (i >= FHW) return;
  float v = __uint_as_float(*vmax) * (float)T_STEPS;
  total[i] = (float)nspkA[i] * (valA[i] + v);
}

// k7: single-block kWTA (8 winners) with feature + 3x3 spatial suppression.
__global__ void k7_kwta(const float* __restrict__ total,
                        float* __restrict__ winners_out) {
  __shared__ float sv[1024];
  __shared__ int   si[1024];
  __shared__ int   wf[8], wr[8], wc[8];
  int tid = threadIdx.x;
  for (int it = 0; it < 8; ++it) {
    float bv = -1.0f; int bi = 0x7fffffff;
    for (int i = tid; i < FHW; i += 1024) {
      int f = i / NSPAT, rm = i % NSPAT, r = rm / HP, c = rm % HP;
      bool sup = false;
      for (int j = 0; j < it; ++j) {
        if (f == wf[j]) { sup = true; break; }
        int dr = r - wr[j]; if (dr < 0) dr = -dr;
        int dc = c - wc[j]; if (dc < 0) dc = -dc;
        if (dr <= 1 && dc <= 1) { sup = true; break; }
      }
      if (sup) continue;
      float v = total[i];
      if (v > bv || (v == bv && i < bi)) { bv = v; bi = i; }
    }
    sv[tid] = bv; si[tid] = bi;
    __syncthreads();
    for (int off = 512; off >= 1; off >>= 1) {
      if (tid < off) {
        float ov = sv[tid + off]; int oi = si[tid + off];
        if (ov > sv[tid] || (ov == sv[tid] && oi < si[tid])) {
          sv[tid] = ov; si[tid] = oi;
        }
      }
      __syncthreads();
    }
    if (tid == 0) {
      float mv = sv[0]; int mi = si[0];
      bool valid = (mv > 0.0f);
      int f = mi / NSPAT, rm = mi % NSPAT, r = rm / HP, c = rm % HP;
      if (valid) { wf[it] = f; wr[it] = r; wc[it] = c; }
      else       { wf[it] = -1; wr[it] = -1000; wc[it] = -1000; f = -1; r = -1; c = -1; }
      winners_out[it * 3 + 0] = (float)f;
      winners_out[it * 3 + 1] = (float)r;
      winners_out[it * 3 + 2] = (float)c;
    }
    __syncthreads();
  }
}

// ---------------------------------------------------------------------------
extern "C" void kernel_launch(void* const* d_in, const int* in_sizes, int n_in,
                              void* d_out, int out_size, void* d_ws, size_t ws_size,
                              hipStream_t stream) {
  (void)in_sizes; (void)n_in; (void)out_size; (void)ws_size;
  const int*   in = (const int*)d_in[0];
  const float* w1 = (const float*)d_in[1];
  const float* w2 = (const float*)d_in[2];

  float* out     = (float*)d_out;
  float* spk_sec = out;                        // (15,250,80,80)
  float* pot_sec = out + (size_t)ELEMS;        // (15,250,80,80)
  float* win_sec = out + 2 * (size_t)ELEMS;    // (8,3)

  char* ws = (char*)d_ws;                      // ~30 MB scratch layout
  _Float16* w2h   = (_Float16*)(ws);                              // 147,456 B
  _Float16* spkp  = (_Float16*)(ws + (160u << 10));               // 5,760,000 B
  int*      maxidx= (int*)(ws + (6u << 20));                      // 384,000 B
  int*      clampf= (int*)(ws + (6u << 20) + (512u << 10));       // 384,000 B
  int*      winch = (int*)(ws + (7u << 20));                      // 25,600 B
  unsigned* vmax  = (unsigned*)(ws + (7u << 20) + (256u << 10));  // 4 B
  float*    valA  = (float*)(ws + (8ull << 20));                  // 6.4 MB
  int*      nspkA = (int*)(ws + (15ull << 20));                   // 6.4 MB
  float*    totalA= (float*)(ws + (22ull << 20));                 // 6.4 MB

  k0_prep<<<(MPAD * KPAD + 255) / 256, 256, 0, stream>>>(w2, w2h, vmax);
  k1_conv1_fire_pool<<<T_STEPS * 25, 256, 0, stream>>>(in, w1, spkp);
  k2_conv2_wmma<<<NTOT / NTILE, 256, 0, stream>>>(spkp, w2h, pot_sec);
  k3_maxarg<<<NTOT / 8, 256, 0, stream>>>(pot_sec, maxidx, clampf);
  k3b_winch<<<(NSPAT + 255) / 256, 256, 0, stream>>>(maxidx, clampf, winch);
  k4_apply<<<(ELEMS + 255) / 256, 256, 0, stream>>>(pot_sec, spk_sec, winch);
  k5_stats<<<(FHW + 255) / 256, 256, 0, stream>>>(pot_sec, spk_sec, valA, nspkA, vmax);
  k6_total<<<(FHW + 255) / 256, 256, 0, stream>>>(valA, nspkA, vmax, totalA);
  k7_kwta<<<1, 1024, 0, stream>>>(totalA, win_sec);
}
